// MyLSTM_35132832482022
// MI455X (gfx1250) — matrix operand order repaired and checked by reading, not compiled, not run on hardware
//
#include <hip/hip_runtime.h>
#include <cstdint>
#include <cstddef>

typedef _Float16 h16_t;
typedef __attribute__((ext_vector_type(8)))  _Float16 v8h;
typedef __attribute__((ext_vector_type(16))) _Float16 v16h;
typedef __attribute__((ext_vector_type(8)))  float    v8f;

#define B_  64
#define T_  512
#define H_  512
#define G4_ 2048            // 4*H
#define M_  (B_ * T_)       // 32768 rows for the input GEMMs

// Build the 16x16x32-f16 A operand from two contiguous 8-half chunks.
// As called (hi = lo + 8): lanes 0-15 hold row M=lane with K = {k0..k0+15},
// lanes 16-31 hold row M=lane-16 with K = {k0+16..k0+31}, matching the B operand.
__device__ __forceinline__ v16h make_a(const h16_t* lo_p, const h16_t* hi_p) {
  v8h lo = *(const v8h*)lo_p;
  v8h hi = *(const v8h*)hi_p;
  v16h a;
#pragma unroll
  for (int i = 0; i < 8; ++i) { a[i] = lo[i]; a[i + 8] = hi[i]; }
  return a;
}

__device__ __forceinline__ v8f vzero8() {
  v8f z;
#pragma unroll
  for (int i = 0; i < 8; ++i) z[i] = 0.0f;
  return z;
}

// ---------------------------------------------------------------------------
// Prep kernels
// ---------------------------------------------------------------------------
__global__ void f32_to_f16_kernel(const float* __restrict__ src,
                                  h16_t* __restrict__ dst, int n) {
  int i = blockIdx.x * blockDim.x + threadIdx.x;
  if (i < n) dst[i] = (h16_t)src[i];
}

// W: [H_][G4_] fp32 row-major  ->  WT: [G4_][H_] f16 row-major (i.e. W^T)
__global__ void transpose_w_kernel(const float* __restrict__ W,
                                   h16_t* __restrict__ WT) {
  int i = blockIdx.x * blockDim.x + threadIdx.x;   // over H_*G4_
  int k = i >> 11;          // row of W   (0..511)
  int n = i & (G4_ - 1);    // col of W   (0..2047)
  WT[(size_t)n * H_ + k] = (h16_t)W[i];
}

// Convert initial h (fp32 [B_*H_]) into the parity-0 half buffer; reset barrier.
__global__ void init_layer_kernel(const float* __restrict__ h0,
                                  h16_t* __restrict__ hbuf0,
                                  int* __restrict__ ctr) {
  int i = blockIdx.x * blockDim.x + threadIdx.x;
  if (i < B_ * H_) hbuf0[i] = (h16_t)h0[i];
  if (i == 0) *ctr = 0;
}

// ---------------------------------------------------------------------------
// Input projection GEMM:  xp[t][b][n] = sum_k A[b*T+t][k] * W[k][n]
// A: [M_][H_] f16 (m = b*T + t), BT: [G4_][H_] f16 (W^T), xp: [T_][B_][G4_] fp32
// One wave computes a 64x16 slab (4 M-tiles, 1 N-tile). Latency hidden by
// occupancy (65536 waves).
// ---------------------------------------------------------------------------
__global__ __launch_bounds__(256) void xproj_gemm_kernel(
    const h16_t* __restrict__ A,
    const h16_t* __restrict__ BT,
    float* __restrict__ xp) {
  const int wave = blockIdx.x * (blockDim.x >> 5) + (threadIdx.x >> 5);
  const int lane = threadIdx.x & 31;
  const int nt   = wave & 127;   // 128 N-tiles of 16
  const int mblk = wave >> 7;    // 512 M-blocks of 64 rows
  const int col  = lane & 15;
  const int hlf  = lane >> 4;

  v8f acc[4];
#pragma unroll
  for (int mi = 0; mi < 4; ++mi) acc[mi] = vzero8();

  // B operand: lane holds column n = nt*16+col, K = [k0+hlf*16, k0+hlf*16+15]
  const h16_t* bRow  = BT + (size_t)(nt * 16 + col) * H_ + hlf * 16;
  const h16_t* aBase = A + (size_t)(mblk * 64 + col) * H_ + hlf * 16;

  // Prologue: operands for ks = 0.
  v16h bv = *(const v16h*)(bRow);
  v16h av[4];
#pragma unroll
  for (int mi = 0; mi < 4; ++mi) {
    const h16_t* ar = aBase + (size_t)mi * 16 * H_;
    av[mi] = make_a(ar, ar + 8);
  }

#pragma unroll
  for (int ks = 0; ks < 16; ++ks) {
    v16h bn;
    v16h an[4];
    if (ks < 15) {              // issue next-iteration loads before computing
      const int k1 = (ks + 1) * 32;
      bn = *(const v16h*)(bRow + k1);
#pragma unroll
      for (int mi = 0; mi < 4; ++mi) {
        const h16_t* ar = aBase + (size_t)mi * 16 * H_ + k1;
        an[mi] = make_a(ar, ar + 8);
      }
    }
#pragma unroll
    for (int mi = 0; mi < 4; ++mi) {
      acc[mi] = __builtin_amdgcn_wmma_f32_16x16x32_f16(
          false, av[mi], false, bv, (short)0, acc[mi], false, false);
    }
    if (ks < 15) {
      bv = bn;
#pragma unroll
      for (int mi = 0; mi < 4; ++mi) av[mi] = an[mi];
    }
  }

  // C/D layout: VGPR r, lane -> (M = r + 8*hlf, N = col)
  const int n = nt * 16 + col;
#pragma unroll
  for (int mi = 0; mi < 4; ++mi) {
#pragma unroll
    for (int r = 0; r < 8; ++r) {
      const int m  = mblk * 64 + mi * 16 + hlf * 8 + r;
      const int bb = m >> 9;        // batch  = m / T_
      const int tt = m & (T_ - 1);  // time   = m % T_
      xp[((size_t)tt * B_ + bb) * G4_ + n] = acc[mi][r];
    }
  }
}

// ---------------------------------------------------------------------------
// Persistent LSTM scan. 32 blocks x 128 threads = 128 waves.
// Block b owns hidden tile jt = b; its 4 waves own batch tiles mt = 0..3.
// The block's Wh slice (4 gates x 16 cols x 512 K, f16 = 64 KB) is preloaded
// into LDS once; every per-step B operand is a ds_load. The A (h) tile is
// burst-loaded in 4-chunk groups with one-burst-ahead prefetch. Cell state c
// stays in registers for all 512 steps; xp gate seeds are double-buffered
// across timesteps. One device-wide barrier per step with proper agent-scope
// release/acquire fences.
// ---------------------------------------------------------------------------
__global__ __launch_bounds__(128) void lstm_scan_kernel(
    const float* __restrict__ xp,    // [T_][B_][G4_]
    const h16_t* __restrict__ WhT,   // [G4_][H_]  (Wh^T)
    h16_t* __restrict__ hbuf,        // 2 * B_*H_ halves (ping/pong)
    const float* __restrict__ c_in,  // [B_*H_]
    const float* __restrict__ bias,  // [8]
    h16_t* __restrict__ y16,         // [M_][H_] (m=b*T+t) or nullptr
    float* __restrict__ yf32,        // [B_][T_][H_] or nullptr
    float* __restrict__ hT_out,      // [B_*H_]
    float* __restrict__ cT_out,      // [B_*H_]
    int* __restrict__ ctr) {
  __shared__ h16_t ldsW[4 * 16 * 512];   // 64 KB: [g][c][k]

  const int lane = threadIdx.x & 31;
  const int mt   = threadIdx.x >> 5;     // 4 batch tiles per block
  const int jt   = blockIdx.x;           // 32 hidden tiles, one per block
  const int col  = lane & 15;
  const int hlf  = lane >> 4;
  const int j    = jt * 16 + col;
  const int rowBase = mt * 16 + hlf * 8; // C-layout rows: rowBase + r

  // ---- one-time: preload this block's Wh slice into LDS ----
  // local row lr = g*16 + c  ->  WhT row n = g*H_ + jt*16 + c
  for (int i = threadIdx.x; i < 4096; i += 128) {  // 4096 chunks of 16 B
    const int lr = i >> 6;                 // 64 chunks per 512-half row
    const int ko = (i & 63) * 8;           // half offset within row
    const int g  = lr >> 4;
    const int cc = lr & 15;
    const h16_t* src = WhT + (size_t)(g * H_ + jt * 16 + cc) * H_ + ko;
    *(v8h*)(ldsW + lr * 512 + ko) = *(const v8h*)src;
  }
  __syncthreads();

  const float bg0 = bias[0] + bias[1];
  const float bg1 = bias[2] + bias[3];
  const float bg2 = bias[4] + bias[5];
  const float bg3 = bias[6] + bias[7];

  // Cell state lives in registers for all 512 steps.
  v8f c;
#pragma unroll
  for (int r = 0; r < 8; ++r) c[r] = c_in[(rowBase + r) * H_ + j];

  // LDS B operand pointers (per gate): lane holds column c=col,
  // K = [k0 + hlf*16, k0 + hlf*16 + 15].
  const h16_t* lB0 = ldsW + (0 * 16 + col) * 512 + hlf * 16;
  const h16_t* lB1 = ldsW + (1 * 16 + col) * 512 + hlf * 16;
  const h16_t* lB2 = ldsW + (2 * 16 + col) * 512 + hlf * 16;
  const h16_t* lB3 = ldsW + (3 * 16 + col) * 512 + hlf * 16;

  const int arow = mt * 16 + col;
  const h16_t* aRowP[2] = {
      hbuf + (size_t)arow * H_ + hlf * 16,                 // parity 0
      hbuf + (size_t)(B_ * H_) + (size_t)arow * H_ + hlf * 16 };

  float hlast[8];
#pragma unroll
  for (int r = 0; r < 8; ++r) hlast[r] = 0.0f;

  // Prime the xp seed double-buffer with t = 0.
  v8f zn0, zn1, zn2, zn3;
  {
    const float* xpt = xp;
#pragma unroll
    for (int r = 0; r < 8; ++r) {
      const size_t base = (size_t)(rowBase + r) * G4_ + j;
      zn0[r] = xpt[base + 0 * H_];
      zn1[r] = xpt[base + 1 * H_];
      zn2[r] = xpt[base + 2 * H_];
      zn3[r] = xpt[base + 3 * H_];
    }
  }

  for (int t = 0; t < T_; ++t) {
    // Consume this step's seeds; immediately issue next step's xp loads so
    // they overlap with the whole WMMA chain + gates + barrier.
    v8f z0 = zn0, z1 = zn1, z2 = zn2, z3 = zn3;
    {
      const int tn = (t < T_ - 1) ? (t + 1) : t;
      const float* xpn = xp + (size_t)tn * B_ * G4_;
#pragma unroll
      for (int r = 0; r < 8; ++r) {
        const size_t base = (size_t)(rowBase + r) * G4_ + j;
        zn0[r] = xpn[base + 0 * H_];
        zn1[r] = xpn[base + 1 * H_];
        zn2[r] = xpn[base + 2 * H_];
        zn3[r] = xpn[base + 3 * H_];
      }
    }

    const h16_t* aRow = aRowP[t & 1];
    h16_t*       hW   = hbuf + (size_t)((t + 1) & 1) * (B_ * H_);

    // A tile in 4 bursts of 4 k-steps, one burst ahead.
    v16h av[4];
#pragma unroll
    for (int ki = 0; ki < 4; ++ki)
      av[ki] = make_a(aRow + ki * 32, aRow + ki * 32 + 8);

#pragma unroll
    for (int kb = 0; kb < 4; ++kb) {
      v16h an[4];
      if (kb < 3) {              // prefetch next burst (global) first
#pragma unroll
        for (int ki = 0; ki < 4; ++ki) {
          const int k1 = (kb + 1) * 128 + ki * 32;
          an[ki] = make_a(aRow + k1, aRow + k1 + 8);
        }
      }
#pragma unroll
      for (int ki = 0; ki < 4; ++ki) {
        const int k0 = kb * 128 + ki * 32;
        v16h b0 = *(const v16h*)(lB0 + k0);   // LDS: low latency
        v16h b1 = *(const v16h*)(lB1 + k0);
        v16h b2 = *(const v16h*)(lB2 + k0);
        v16h b3 = *(const v16h*)(lB3 + k0);
        z0 = __builtin_amdgcn_wmma_f32_16x16x32_f16(false, av[ki], false, b0, (short)0, z0, false, false);
        z1 = __builtin_amdgcn_wmma_f32_16x16x32_f16(false, av[ki], false, b1, (short)0, z1, false, false);
        z2 = __builtin_amdgcn_wmma_f32_16x16x32_f16(false, av[ki], false, b2, (short)0, z2, false, false);
        z3 = __builtin_amdgcn_wmma_f32_16x16x32_f16(false, av[ki], false, b3, (short)0, z3, false, false);
      }
      if (kb < 3) {
#pragma unroll
        for (int ki = 0; ki < 4; ++ki) av[ki] = an[ki];
      }
    }

    // Gate fusion + state update (all in registers).
#pragma unroll
    for (int r = 0; r < 8; ++r) {
      const float ig = 1.0f / (1.0f + __expf(-(z0[r] + bg0)));
      const float fg = 1.0f / (1.0f + __expf(-(z1[r] + bg1)));
      const float gg = tanhf(z2[r] + bg2);
      const float og = 1.0f / (1.0f + __expf(-(z3[r] + bg3)));
      const float cn = fg * c[r] + ig * gg;
      c[r] = cn;
      const float hn = og * tanhf(cn);
      hlast[r] = hn;
      const int row = rowBase + r;
      hW[row * H_ + j] = (h16_t)hn;
      if (y16)  y16[((size_t)row * T_ + t) * H_ + j] = (h16_t)hn;   // m = b*T + t
      if (yf32) yf32[((size_t)row * T_ + t) * H_ + j] = hn;         // [B][T][H]
    }

    // Device-wide barrier: 32 blocks, monotone counter (no reset races).
    __builtin_amdgcn_fence(__ATOMIC_RELEASE, "agent");   // publish h stores
    __syncthreads();
    if (threadIdx.x == 0) {
      __hip_atomic_fetch_add(ctr, 1, __ATOMIC_RELEASE, __HIP_MEMORY_SCOPE_AGENT);
      const int target = (int)gridDim.x * (t + 1);
      while (__hip_atomic_load(ctr, __ATOMIC_ACQUIRE, __HIP_MEMORY_SCOPE_AGENT) < target) {
        __builtin_amdgcn_s_sleep(1);
      }
    }
    __syncthreads();
    __builtin_amdgcn_fence(__ATOMIC_ACQUIRE, "agent");   // invalidate stale h
  }

  // Final h/c from registers.
#pragma unroll
  for (int r = 0; r < 8; ++r) {
    const int row = rowBase + r;
    hT_out[row * H_ + j] = hlast[r];
    cT_out[row * H_ + j] = c[r];
  }
}

// ---------------------------------------------------------------------------
// Host launcher
// ---------------------------------------------------------------------------
extern "C" void kernel_launch(void* const* d_in, const int* in_sizes, int n_in,
                              void* d_out, int out_size, void* d_ws, size_t ws_size,
                              hipStream_t stream) {
  (void)in_sizes; (void)n_in; (void)out_size; (void)ws_size;
  const float* x   = (const float*)d_in[0];
  const float* h   = (const float*)d_in[1];   // [2][B][H]
  const float* c   = (const float*)d_in[2];   // [2][B][H]
  const float* Wx0 = (const float*)d_in[3];
  const float* Wh0 = (const float*)d_in[4];
  const float* b0  = (const float*)d_in[5];
  const float* Wx1 = (const float*)d_in[6];
  const float* Wh1 = (const float*)d_in[7];
  const float* b1  = (const float*)d_in[8];
  float* out = (float*)d_out;

  // d_out layout: out1 [B,T,H] | new_h [2,B,H] | new_c [2,B,H]
  float* out_y  = out;
  float* out_h  = out + (size_t)B_ * T_ * H_;
  float* out_c  = out_h + (size_t)2 * B_ * H_;

  // Workspace layout (all offsets well-aligned).
  char* ws = (char*)d_ws;
  size_t off = 0;
  float* xp   = (float*)(ws + off); off += (size_t)T_ * B_ * G4_ * sizeof(float); // 256 MB
  h16_t* x16  = (h16_t*)(ws + off); off += (size_t)M_ * H_ * sizeof(h16_t);       // 32 MB
  h16_t* y016 = (h16_t*)(ws + off); off += (size_t)M_ * H_ * sizeof(h16_t);       // 32 MB
  const size_t wbytes = (size_t)G4_ * H_ * sizeof(h16_t);                         // 2 MB each
  h16_t* WxT0 = (h16_t*)(ws + off); off += wbytes;
  h16_t* WhT0 = (h16_t*)(ws + off); off += wbytes;
  h16_t* WxT1 = (h16_t*)(ws + off); off += wbytes;
  h16_t* WhT1 = (h16_t*)(ws + off); off += wbytes;
  h16_t* hbuf = (h16_t*)(ws + off); off += (size_t)2 * B_ * H_ * sizeof(h16_t);   // ping/pong
  int*   ctr0 = (int*)(ws + off);   off += 256;
  int*   ctr1 = (int*)(ws + off);   off += 256;

  // ---- shared prep ----
  f32_to_f16_kernel<<<(M_ * H_ + 255) / 256, 256, 0, stream>>>(x, x16, M_ * H_);
  transpose_w_kernel<<<(H_ * G4_) / 256, 256, 0, stream>>>(Wx0, WxT0);
  transpose_w_kernel<<<(H_ * G4_) / 256, 256, 0, stream>>>(Wh0, WhT0);
  transpose_w_kernel<<<(H_ * G4_) / 256, 256, 0, stream>>>(Wx1, WxT1);
  transpose_w_kernel<<<(H_ * G4_) / 256, 256, 0, stream>>>(Wh1, WhT1);

  const int gemm_blocks = (M_ / 64) * (G4_ / 16) / 8;  // 65536 waves / 8 per block

  // ---- layer 0 ----
  init_layer_kernel<<<(B_ * H_ + 255) / 256, 256, 0, stream>>>(h, hbuf, ctr0);
  xproj_gemm_kernel<<<gemm_blocks, 256, 0, stream>>>(x16, WxT0, xp);
  lstm_scan_kernel<<<32, 128, 0, stream>>>(xp, WhT0, hbuf, c, b0,
                                           y016, nullptr,
                                           out_h, out_c, ctr0);

  // ---- layer 1 ----
  init_layer_kernel<<<(B_ * H_ + 255) / 256, 256, 0, stream>>>(h + B_ * H_, hbuf, ctr1);
  xproj_gemm_kernel<<<gemm_blocks, 256, 0, stream>>>(y016, WxT1, xp);
  lstm_scan_kernel<<<32, 128, 0, stream>>>(xp, WhT1, hbuf, c + B_ * H_, b1,
                                           nullptr, out_y,
                                           out_h + B_ * H_, out_c + B_ * H_, ctr1);
}
